// LMBlock_44461501448900
// MI455X (gfx1250) — matrix operand order repaired and checked by reading, not compiled, not run on hardware
//
#include <hip/hip_runtime.h>

// ---------------------------------------------------------------------------
// MI455X (gfx1250, wave32) DeltaNet block.
// All GEMMs: v_wmma_f32_16x16x32_bf16 (bf16 in, f32 accum).
// GEMM tile staging: GLOBAL_LOAD_ASYNC_TO_LDS_B128 (ASYNCcnt ping-pong).
// Delta recurrence: chunked (C=32) formulation -> WMMA GEMMs + tiny solve,
// state S kept f32 in LDS with a transposed bf16 shadow for WMMA B-operand.
// ---------------------------------------------------------------------------

typedef __bf16 bf16;
typedef __attribute__((ext_vector_type(16))) __bf16 v16bf;
typedef __attribute__((ext_vector_type(8)))  float  v8f;
typedef __attribute__((ext_vector_type(4)))  float  f32x4;   // 16-byte chunk

constexpr int Bc  = 2;
constexpr int Tc  = 2048;
constexpr int Dc  = 2048;
constexpr int Hc  = 16;
constexpr int HDc = 128;
constexpr int Ic  = 5632;
constexpr float EPSc = 1e-6f;
constexpr int ROWSc = Bc * Tc;          // 4096 tokens

// Async global->LDS (GV mode): vdst = LDS byte address, vaddr = 64-bit global.
// Tracked by ASYNCcnt; completion via s_wait_asynccnt.
#define ASYNC_LDS_B128(ldsoff, gptr)                                        \
  asm volatile("global_load_async_to_lds_b128 %0, %1, off"                  \
               :: "v"(ldsoff), "v"((unsigned long long)(size_t)(gptr))      \
               : "memory")
#define WAIT_ASYNC0() asm volatile("s_wait_asynccnt 0x0" ::: "memory")

__device__ inline unsigned lds_addr_of(const void* p) {
  return (unsigned)(size_t)p;   // low 32 bits of a shared generic pointer
}

__device__ inline v8f wmma_bf16(v16bf a, v16bf b, v8f c) {
  // D = A(16x32 bf16) * B(32x16 bf16) + C(16x16 f32)
  return __builtin_amdgcn_wmma_f32_16x16x32_bf16(
      /*neg_a=*/false, a, /*neg_b=*/false, b,
      /*c_mod=*/(short)0, c, /*reuse_a=*/false, /*reuse_b=*/false);
}

struct BFrag { union { v16bf v; f32x4 q[2]; }; };

// A-operand fragment: tile points at (row0, k0) of row-major [rows][stride].
// Lane L: row = L%16, K = {8*(L>>4)..+7} in q[0], {+16..+23} in q[1].
__device__ inline v16bf ldsA(const bf16* tile, int stride, int lane) {
  int r = lane & 15, b = (lane >> 4) << 3;
  BFrag f;
  f.q[0] = *(const f32x4*)(tile + r * stride + b);
  f.q[1] = *(const f32x4*)(tile + r * stride + b + 16);
  return f.v;
}

// B-operand fragment: tile points at (n0, k0) of [n][k] (column-major B).
// Lane L: n = L%16, K = {8*(L>>4)..+7} in q[0], {+16..+23} in q[1].
__device__ inline v16bf ldsB(const bf16* tile, int stride, int lane) {
  int n = lane & 15, b = (lane >> 4) << 3;
  BFrag f;
  f.q[0] = *(const f32x4*)(tile + n * stride + b);
  f.q[1] = *(const f32x4*)(tile + n * stride + b + 16);
  return f.v;
}

// ---------------------------------------------------------------------------
// f32 [R][C] -> bf16 [C][R]  (weights become [N][K] for the WMMA B-operand)
// ---------------------------------------------------------------------------
__global__ __launch_bounds__(256)
void cast_transpose(const float* __restrict__ in, bf16* __restrict__ out,
                    int R, int C) {
  __shared__ bf16 tile[32][33];
  int c0 = blockIdx.x * 32, r0 = blockIdx.y * 32;
  int tx = threadIdx.x & 31, ty = threadIdx.x >> 5;   // ty in 0..7
#pragma unroll
  for (int i = 0; i < 4; ++i) {
    int r = r0 + ty + i * 8, c = c0 + tx;
    if (r < R && c < C) tile[ty + i * 8][tx] = (bf16)in[(size_t)r * C + c];
  }
  __syncthreads();
#pragma unroll
  for (int i = 0; i < 4; ++i) {
    int c = c0 + ty + i * 8, r = r0 + tx;
    if (r < R && c < C) out[(size_t)c * R + r] = tile[tx][ty + i * 8];
  }
}

// ---------------------------------------------------------------------------
// RMSNorm over D=2048: f32 in -> bf16 out
// ---------------------------------------------------------------------------
__global__ __launch_bounds__(256)
void rmsnorm_kernel(const float* __restrict__ x, const float* __restrict__ w,
                    bf16* __restrict__ out) {
  int row = blockIdx.x;
  const float* xr = x + (size_t)row * Dc;
  __shared__ float red[256];
  float ss = 0.f;
  for (int i = threadIdx.x; i < Dc; i += 256) { float v = xr[i]; ss += v * v; }
  red[threadIdx.x] = ss;
  __syncthreads();
  for (int off = 128; off; off >>= 1) {
    if (threadIdx.x < off) red[threadIdx.x] += red[threadIdx.x + off];
    __syncthreads();
  }
  float scale = rsqrtf(red[0] * (1.f / Dc) + EPSc);
  bf16* orow = out + (size_t)row * Dc;
  for (int i = threadIdx.x; i < Dc; i += 256)
    orow[i] = (bf16)(xr[i] * scale * w[i]);
}

// ---------------------------------------------------------------------------
// Per-128-segment norm. RMS=true: rmsnorm with weight; RMS=false: l2norm.
// ---------------------------------------------------------------------------
template <bool RMS, typename Tin>
__global__ __launch_bounds__(128)
void seg_norm(const Tin* __restrict__ in, const float* __restrict__ w,
              bf16* __restrict__ out) {
  size_t base = (size_t)blockIdx.x * 128;
  __shared__ float red[128];
  float v = (float)in[base + threadIdx.x];
  red[threadIdx.x] = v * v;
  __syncthreads();
  for (int off = 64; off; off >>= 1) {
    if (threadIdx.x < off) red[threadIdx.x] += red[threadIdx.x + off];
    __syncthreads();
  }
  float ss = red[0];
  float scale = RMS ? rsqrtf(ss * (1.f / 128.f) + EPSc) : rsqrtf(ss + EPSc);
  float o = v * scale;
  if (RMS) o *= w[threadIdx.x];
  out[base + threadIdx.x] = (bf16)o;
}

// ---------------------------------------------------------------------------
// beta = sigmoid(xn @ Wb): block per token row, 16 threads per head.
// ---------------------------------------------------------------------------
__global__ __launch_bounds__(256)
void beta_kernel(const bf16* __restrict__ xn, const bf16* __restrict__ WbT,
                 float* __restrict__ beta) {
  int row = blockIdx.x;
  __shared__ bf16 xrow[Dc];
  __shared__ float red[256];
  for (int i = threadIdx.x; i < Dc; i += 256)
    xrow[i] = xn[(size_t)row * Dc + i];
  __syncthreads();
  int h = threadIdx.x >> 4, s = threadIdx.x & 15;
  float sum = 0.f;
  const int seg = Dc / 16;                 // 128
  const bf16* wr = WbT + (size_t)h * Dc + s * seg;
  const bf16* xs = xrow + s * seg;
#pragma unroll 8
  for (int j = 0; j < seg; ++j) sum += (float)xs[j] * (float)wr[j];
  red[threadIdx.x] = sum;
  __syncthreads();
  for (int off = 8; off; off >>= 1) {
    if (s < off) red[threadIdx.x] += red[threadIdx.x + off];
    __syncthreads();
  }
  if (s == 0)
    beta[(size_t)row * Hc + h] = 1.f / (1.f + __expf(-red[threadIdx.x]));
}

// ---------------------------------------------------------------------------
// bf16 WMMA GEMM: C[M][N] = epilogue(A[M][K] @ Bt[N][K]^T)
// Block = 256 thr (8 waves), tile 128x128, BK=32.
// Ping-pong LDS buffers filled by GLOBAL_LOAD_ASYNC_TO_LDS_B128: the next
// tile's fill (ASYNCcnt) overlaps the current 8-WMMA compute; s_wait_asynccnt
// lands after the math, just before the barrier.
// EPI: 0 f32 | 1 silu->bf16 | 2 bf16 | 3 +resid->f32 | 4 aux(silu-gate)*acc->bf16
// ---------------------------------------------------------------------------
template <int EPI>
__global__ __launch_bounds__(256)
void gemm_bf16_kernel(const bf16* __restrict__ A, const bf16* __restrict__ Bt,
                      void* __restrict__ Out, const float* __restrict__ resid,
                      const bf16* __restrict__ aux, int M, int N, int K) {
  __shared__ bf16 As[2][128 * 32];
  __shared__ bf16 Bs[2][128 * 32];
  int tid = threadIdx.x, lane = tid & 31, w = tid >> 5;
  int m0 = blockIdx.y * 128, n0 = blockIdx.x * 128;
  int wm = (w >> 2) * 64, wn = (w & 3) * 32;   // wave tile: 64 rows x 32 cols
  int lr = tid >> 1, lc = (tid & 1) * 16;      // each thread stages 16 bf16

  const bf16* Ag = A + (size_t)(m0 + lr) * K + lc;
  const bf16* Bg = Bt + (size_t)(n0 + lr) * K + lc;
  unsigned la[2], lb[2];
  la[0] = lds_addr_of(&As[0][lr * 32 + lc]);
  la[1] = lds_addr_of(&As[1][lr * 32 + lc]);
  lb[0] = lds_addr_of(&Bs[0][lr * 32 + lc]);
  lb[1] = lds_addr_of(&Bs[1][lr * 32 + lc]);

  v8f acc[4][2];
#pragma unroll
  for (int mi = 0; mi < 4; ++mi)
#pragma unroll
    for (int ni = 0; ni < 2; ++ni) acc[mi][ni] = (v8f)0.0f;

  int nk = K >> 5;
  // prologue: fill buffer 0
  ASYNC_LDS_B128(la[0],      Ag);
  ASYNC_LDS_B128(la[0] + 16, Ag + 8);
  ASYNC_LDS_B128(lb[0],      Bg);
  ASYNC_LDS_B128(lb[0] + 16, Bg + 8);
  WAIT_ASYNC0();
  __syncthreads();

#pragma unroll 1
  for (int kt = 0; kt < nk; ++kt) {
    int cur = kt & 1;
    if (kt + 1 < nk) {                        // fire next tile, don't wait
      int k0 = (kt + 1) << 5, nb = cur ^ 1;
      ASYNC_LDS_B128(la[nb],      Ag + k0);
      ASYNC_LDS_B128(la[nb] + 16, Ag + k0 + 8);
      ASYNC_LDS_B128(lb[nb],      Bg + k0);
      ASYNC_LDS_B128(lb[nb] + 16, Bg + k0 + 8);
    }
    v16bf af[4], bfr[2];
#pragma unroll
    for (int mi = 0; mi < 4; ++mi)
      af[mi] = ldsA(&As[cur][(wm + mi * 16) * 32], 32, lane);
#pragma unroll
    for (int ni = 0; ni < 2; ++ni)
      bfr[ni] = ldsB(&Bs[cur][(wn + ni * 16) * 32], 32, lane);
#pragma unroll
    for (int mi = 0; mi < 4; ++mi)
#pragma unroll
      for (int ni = 0; ni < 2; ++ni)
        acc[mi][ni] = wmma_bf16(af[mi], bfr[ni], acc[mi][ni]);
    WAIT_ASYNC0();                            // hidden behind the 8 WMMAs
    __syncthreads();
  }

  int rbase = m0 + wm + ((lane >> 4) << 3);
  int cbase = n0 + wn + (lane & 15);
#pragma unroll
  for (int mi = 0; mi < 4; ++mi)
#pragma unroll
    for (int ni = 0; ni < 2; ++ni)
#pragma unroll
      for (int r = 0; r < 8; ++r) {
        size_t idx = (size_t)(rbase + mi * 16 + r) * N + (cbase + ni * 16);
        float v = acc[mi][ni][r];
        if (EPI == 0)      ((float*)Out)[idx] = v;
        else if (EPI == 1) ((bf16*)Out)[idx] = (bf16)(v / (1.f + __expf(-v)));
        else if (EPI == 2) ((bf16*)Out)[idx] = (bf16)v;
        else if (EPI == 3) ((float*)Out)[idx] = v + resid[idx];
        else               ((bf16*)Out)[idx] = (bf16)(v * (float)aux[idx]);
      }
}

// ---------------------------------------------------------------------------
// Chunked delta rule, C=32. One workgroup (8 waves) per (b,h).
// Per chunk (all WMMA except a barrier-free 128-lane forward substitution):
//   A = Kc @ S0 ; M = Kc Kc^T ; P = tril(Qc Kc^T)
//   delta_i = beta_i (v_i - A_i - sum_{j<i} M_ij delta_j)
//   S += Kc^T Delta ; O = Qc @ S0 + P @ Delta
// ---------------------------------------------------------------------------
__global__ __launch_bounds__(256)
void delta_kernel(const bf16* __restrict__ qg, const bf16* __restrict__ kg,
                  const bf16* __restrict__ vg, const float* __restrict__ betag,
                  float* __restrict__ og, float* __restrict__ Sg) {
  const int bh = blockIdx.x, b = bh >> 4, h = bh & 15;
  __shared__ float S[128 * 128];    // master state, S[k*128+v]  (64 KB)
  __shared__ bf16  Sb[128 * 128];   // bf16 shadow, Sb[v*128+k]  (32 KB)
  __shared__ bf16  Kc[32 * 128];    // chunk K, [t][k]
  __shared__ bf16  KcT[128 * 32];   // chunk K^T, [k][t]
  __shared__ bf16  Qc[32 * 128];    // chunk Q, [t][k]
  __shared__ float Wm[32 * 128];    // V - A - updates, [t][v]
  __shared__ float Mm[32 * 32];     // K K^T
  __shared__ bf16  Pm[32 * 32];     // masked Q K^T
  __shared__ bf16  DbT[128 * 32];   // Delta^T, [v][t]
  __shared__ float betas[32];

  int tid = threadIdx.x, lane = tid & 31, w = tid >> 5;

  for (int i = tid; i < 128 * 128; i += 256) { S[i] = 0.f; Sb[i] = (bf16)0.0f; }
  __syncthreads();

#pragma unroll 1
  for (int c = 0; c < Tc / 32; ++c) {
    // ---- stage chunk into LDS ----
    {
      int tt = tid >> 3, off = (tid & 7) * 16;
      size_t gbase = ((size_t)(b * Tc + c * 32 + tt) * Hc + h) * 128 + off;
      // Q goes straight to LDS via the async path (no transform needed)
      unsigned lq = lds_addr_of(&Qc[tt * 128 + off]);
      ASYNC_LDS_B128(lq,      qg + gbase);
      ASYNC_LDS_B128(lq + 16, qg + gbase + 8);
      union { f32x4 q[2]; bf16 e[16]; } tk, tv;
      tk.q[0] = *(const f32x4*)(kg + gbase);
      tk.q[1] = *(const f32x4*)(kg + gbase + 8);
      *(f32x4*)&Kc[tt * 128 + off] = tk.q[0];
      *(f32x4*)&Kc[tt * 128 + off + 8] = tk.q[1];
#pragma unroll
      for (int j = 0; j < 16; ++j) KcT[(off + j) * 32 + tt] = tk.e[j];
      tv.q[0] = *(const f32x4*)(vg + gbase);
      tv.q[1] = *(const f32x4*)(vg + gbase + 8);
#pragma unroll
      for (int j = 0; j < 16; ++j) Wm[tt * 128 + off + j] = (float)tv.e[j];
      if (tid < 32)
        betas[tid] = betag[(size_t)(b * Tc + c * 32 + tid) * Hc + h];
    }
    WAIT_ASYNC0();
    __syncthreads();

    // ---- G1: Wm -= Kc@S0 ; Mm = Kc Kc^T ; Pm = tril(Qc Kc^T) ----
#pragma unroll
    for (int rep = 0; rep < 2; ++rep) {
      int tile = w + rep * 8, tm = tile >> 3, tn = tile & 7;
      v8f acc = (v8f)0.0f;
#pragma unroll
      for (int kk = 0; kk < 4; ++kk) {
        v16bf a = ldsA(&Kc[tm * 16 * 128 + kk * 32], 128, lane);
        v16bf bb = ldsB(&Sb[(tn * 16) * 128 + kk * 32], 128, lane);
        acc = wmma_bf16(a, bb, acc);
      }
      int r0 = tm * 16 + ((lane >> 4) << 3), cv = tn * 16 + (lane & 15);
#pragma unroll
      for (int r = 0; r < 8; ++r) Wm[(r0 + r) * 128 + cv] -= acc[r];
    }
    if (w < 4) {                       // wave-uniform branch; EXEC stays full
      int tm = w >> 1, tn = w & 1;
      v8f acc = (v8f)0.0f;
#pragma unroll
      for (int kk = 0; kk < 4; ++kk) {
        v16bf a = ldsA(&Kc[tm * 16 * 128 + kk * 32], 128, lane);
        v16bf bb = ldsB(&Kc[(tn * 16) * 128 + kk * 32], 128, lane);
        acc = wmma_bf16(a, bb, acc);
      }
      int r0 = tm * 16 + ((lane >> 4) << 3), cj = tn * 16 + (lane & 15);
#pragma unroll
      for (int r = 0; r < 8; ++r) Mm[(r0 + r) * 32 + cj] = acc[r];
    } else {
      int ww = w - 4, tm = ww >> 1, tn = ww & 1;
      v8f acc = (v8f)0.0f;
#pragma unroll
      for (int kk = 0; kk < 4; ++kk) {
        v16bf a = ldsA(&Qc[tm * 16 * 128 + kk * 32], 128, lane);
        v16bf bb = ldsB(&Kc[(tn * 16) * 128 + kk * 32], 128, lane);
        acc = wmma_bf16(a, bb, acc);
      }
      int r0 = tm * 16 + ((lane >> 4) << 3), cj = tn * 16 + (lane & 15);
#pragma unroll
      for (int r = 0; r < 8; ++r) {
        int i = r0 + r;
        Pm[i * 32 + cj] = (bf16)((cj <= i) ? acc[r] : 0.f);  // incl. diagonal
      }
    }
    __syncthreads();

    // ---- forward substitution: each of 128 lanes owns one v-column ----
    if (tid < 128) {
      int v = tid;
      float wc[32];
#pragma unroll
      for (int i = 0; i < 32; ++i) wc[i] = Wm[i * 128 + v];
#pragma unroll 1
      for (int j = 0; j < 32; ++j) {
        float dj = betas[j] * wc[j];
        DbT[v * 32 + j] = (bf16)dj;
        for (int i = j + 1; i < 32; ++i) wc[i] -= Mm[i * 32 + j] * dj;
      }
    }
    __syncthreads();

    // ---- G2: O = Qc@S0 + Pm@Delta (global) ; S += Kc^T Delta (LDS) ----
#pragma unroll
    for (int rep = 0; rep < 2; ++rep) {
      int tile = w + rep * 8, tm = tile >> 3, tn = tile & 7;
      v8f acc = (v8f)0.0f;
#pragma unroll
      for (int kk = 0; kk < 4; ++kk) {
        v16bf a = ldsA(&Qc[tm * 16 * 128 + kk * 32], 128, lane);
        v16bf bb = ldsB(&Sb[(tn * 16) * 128 + kk * 32], 128, lane);
        acc = wmma_bf16(a, bb, acc);
      }
      {
        v16bf a = ldsA(&Pm[tm * 16 * 32], 32, lane);
        v16bf bb = ldsB(&DbT[(tn * 16) * 32], 32, lane);
        acc = wmma_bf16(a, bb, acc);
      }
      int r0 = tm * 16 + ((lane >> 4) << 3), cv = tn * 16 + (lane & 15);
#pragma unroll
      for (int r = 0; r < 8; ++r) {
        int tt = r0 + r;
        og[((size_t)(b * Tc + c * 32 + tt) * Hc + h) * 128 + cv] = acc[r];
      }
    }
#pragma unroll 1
    for (int vn = 0; vn < 8; ++vn) {
      int km = w;
      int r0 = km * 16 + ((lane >> 4) << 3), cv = vn * 16 + (lane & 15);
      v8f acc;
#pragma unroll
      for (int r = 0; r < 8; ++r) acc[r] = S[(r0 + r) * 128 + cv];
      v16bf a = ldsA(&KcT[(km * 16) * 32], 32, lane);
      v16bf bb = ldsB(&DbT[(vn * 16) * 32], 32, lane);
      acc = wmma_bf16(a, bb, acc);
#pragma unroll
      for (int r = 0; r < 8; ++r) S[(r0 + r) * 128 + cv] = acc[r];
    }
    __syncthreads();

    // ---- refresh bf16 shadow (transposed) ----
    for (int e = tid; e < 128 * 128; e += 256) {
      int kk = e >> 7, vv = e & 127;
      Sb[vv * 128 + kk] = (bf16)S[e];
    }
    __syncthreads();
  }

  for (int e = tid; e < 128 * 128; e += 256)
    Sg[(size_t)bh * 128 * 128 + e] = S[e];
}

// ---------------------------------------------------------------------------
// Host orchestration
// ---------------------------------------------------------------------------
extern "C" void kernel_launch(void* const* d_in, const int* in_sizes, int n_in,
                              void* d_out, int out_size, void* d_ws,
                              size_t ws_size, hipStream_t stream) {
  (void)in_sizes; (void)n_in; (void)out_size; (void)ws_size;
  const float* x   = (const float*)d_in[0];
  const float* anw = (const float*)d_in[1];
  const float* Wq  = (const float*)d_in[2];
  const float* Wk  = (const float*)d_in[3];
  const float* Wv  = (const float*)d_in[4];
  const float* Wb  = (const float*)d_in[5];
  const float* onw = (const float*)d_in[6];
  const float* Wo  = (const float*)d_in[7];
  const float* mnw = (const float*)d_in[8];
  const float* Wg  = (const float*)d_in[9];
  const float* Wu  = (const float*)d_in[10];
  const float* Wd  = (const float*)d_in[11];

  float* out_x = (float*)d_out;
  float* out_S = out_x + (size_t)Bc * Tc * Dc;

  char* p = (char*)d_ws;
  auto alloc = [&](size_t bytes) -> void* {
    void* r = (void*)p;
    p += (bytes + 255) & ~(size_t)255;
    return r;
  };
  bf16* WqT = (bf16*)alloc((size_t)Dc * Dc * 2);
  bf16* WkT = (bf16*)alloc((size_t)Dc * Dc * 2);
  bf16* WvT = (bf16*)alloc((size_t)Dc * Dc * 2);
  bf16* WoT = (bf16*)alloc((size_t)Dc * Dc * 2);
  bf16* WgT = (bf16*)alloc((size_t)Ic * Dc * 2);
  bf16* WuT = (bf16*)alloc((size_t)Ic * Dc * 2);
  bf16* WdT = (bf16*)alloc((size_t)Dc * Ic * 2);
  bf16* WbT = (bf16*)alloc((size_t)Hc * Dc * 2);
  bf16* xn  = (bf16*)alloc((size_t)ROWSc * Dc * 2);
  bf16* qb  = (bf16*)alloc((size_t)ROWSc * Dc * 2);
  bf16* kb  = (bf16*)alloc((size_t)ROWSc * Dc * 2);
  bf16* vb  = (bf16*)alloc((size_t)ROWSc * Dc * 2);
  float* beta = (float*)alloc((size_t)ROWSc * Hc * 4);
  float* o    = (float*)alloc((size_t)ROWSc * Dc * 4);
  bf16* on    = (bf16*)alloc((size_t)ROWSc * Dc * 2);
  float* x1   = (float*)alloc((size_t)ROWSc * Dc * 4);
  bf16* mn    = (bf16*)alloc((size_t)ROWSc * Dc * 2);
  bf16* Gs    = (bf16*)alloc((size_t)ROWSc * Ic * 2);
  bf16* hb    = (bf16*)alloc((size_t)ROWSc * Ic * 2);

  // --- weight bf16 cast + transpose ([K][N] f32 -> [N][K] bf16) ---
  dim3 tb(256);
  cast_transpose<<<dim3(Dc / 32, Dc / 32), tb, 0, stream>>>(Wq, WqT, Dc, Dc);
  cast_transpose<<<dim3(Dc / 32, Dc / 32), tb, 0, stream>>>(Wk, WkT, Dc, Dc);
  cast_transpose<<<dim3(Dc / 32, Dc / 32), tb, 0, stream>>>(Wv, WvT, Dc, Dc);
  cast_transpose<<<dim3(Dc / 32, Dc / 32), tb, 0, stream>>>(Wo, WoT, Dc, Dc);
  cast_transpose<<<dim3(Ic / 32, Dc / 32), tb, 0, stream>>>(Wg, WgT, Dc, Ic);
  cast_transpose<<<dim3(Ic / 32, Dc / 32), tb, 0, stream>>>(Wu, WuT, Dc, Ic);
  cast_transpose<<<dim3(Dc / 32, Ic / 32), tb, 0, stream>>>(Wd, WdT, Ic, Dc);
  cast_transpose<<<dim3(1, Dc / 32), tb, 0, stream>>>(Wb, WbT, Dc, Hc);

  // --- attention sublayer ---
  rmsnorm_kernel<<<ROWSc, 256, 0, stream>>>(x, anw, xn);

  dim3 gD(Dc / 128, ROWSc / 128);       // N=2048 tiles
  dim3 gI(Ic / 128, ROWSc / 128);       // N=5632 tiles
  gemm_bf16_kernel<1><<<gD, 256, 0, stream>>>(xn, WqT, qb, nullptr, nullptr,
                                              ROWSc, Dc, Dc);  // silu
  gemm_bf16_kernel<1><<<gD, 256, 0, stream>>>(xn, WkT, kb, nullptr, nullptr,
                                              ROWSc, Dc, Dc);  // silu
  gemm_bf16_kernel<2><<<gD, 256, 0, stream>>>(xn, WvT, vb, nullptr, nullptr,
                                              ROWSc, Dc, Dc);  // raw bf16
  beta_kernel<<<ROWSc, 256, 0, stream>>>(xn, WbT, beta);

  seg_norm<false, bf16><<<ROWSc * Hc, 128, 0, stream>>>(qb, nullptr, qb);
  seg_norm<false, bf16><<<ROWSc * Hc, 128, 0, stream>>>(kb, nullptr, kb);

  delta_kernel<<<Bc * Hc, 256, 0, stream>>>(qb, kb, vb, beta, o, out_S);

  seg_norm<true, float><<<ROWSc * Hc, 128, 0, stream>>>(o, onw, on);
  gemm_bf16_kernel<3><<<gD, 256, 0, stream>>>(on, WoT, x1, x, nullptr,
                                              ROWSc, Dc, Dc);  // + residual

  // --- SwiGLU MLP sublayer ---
  rmsnorm_kernel<<<ROWSc, 256, 0, stream>>>(x1, mnw, mn);
  gemm_bf16_kernel<1><<<gI, 256, 0, stream>>>(mn, WgT, Gs, nullptr, nullptr,
                                              ROWSc, Ic, Dc);  // silu gate
  gemm_bf16_kernel<4><<<gI, 256, 0, stream>>>(mn, WuT, hb, nullptr, Gs,
                                              ROWSc, Ic, Dc);  // gate * up
  gemm_bf16_kernel<3><<<gD, 256, 0, stream>>>(hb, WdT, out_x, x1, nullptr,
                                              ROWSc, Dc, Ic);  // + residual
}
